// RouterCond_27195732918429
// MI455X (gfx1250) — matrix operand order repaired and checked by reading, not compiled, not run on hardware
//
#include <hip/hip_runtime.h>
#include <hip/hip_bf16.h>

// RouterCond fused kernel for MI455X (gfx1250, wave32, WMMA bf16 + TDM staging).
// d_out layout (flat, float): router_mask[N,64] | top_i[N,2] | router_probs[N,64] | probs[N,64]

#define DIN   1536   // D + DC
#define D0    1024   // inputs feature dim
#define DC0   512    // cond feature dim
#define HDIM  3072   // 2*din
#define NEXP  64     // experts
#define TPB   16     // tokens per block

typedef __bf16 bf16;
typedef bf16  v16bf __attribute__((ext_vector_type(16)));
typedef bf16  v8bf  __attribute__((ext_vector_type(8)));
typedef float v8f   __attribute__((ext_vector_type(8)));

typedef unsigned int u32x4 __attribute__((ext_vector_type(4)));
typedef int          i32x8 __attribute__((ext_vector_type(8)));
typedef int          i32x4 __attribute__((ext_vector_type(4)));

union V16U { v16bf v; v8bf h[2]; };

#if defined(__has_builtin)
#if __has_builtin(__builtin_amdgcn_tensor_load_to_lds)
#define HAVE_TDM 1
#endif
#endif

#ifdef HAVE_TDM
// Generic pointer -> 32-bit LDS byte offset (addrspacecast strips the aperture).
static __device__ __forceinline__ unsigned int lds_byte_offset(const void* p) {
    return (unsigned int)(size_t)(__attribute__((address_space(3))) const void*)p;
}
#endif

// ---------------------------------------------------------------------------
// Kernel 0: build bf16 operands in workspace:
//   Xb[N,1536] = bf16(concat(X, cond)),  W1b = bf16(W1),  W2b = bf16(W2)
// ---------------------------------------------------------------------------
__global__ __launch_bounds__(256)
void prep_kernel(const float* __restrict__ X, const float* __restrict__ C,
                 const float* __restrict__ W1, const float* __restrict__ W2,
                 bf16* __restrict__ Xb, bf16* __restrict__ W1b, bf16* __restrict__ W2b,
                 int Ntok, int n1, int n2) {
    const size_t stride = (size_t)gridDim.x * blockDim.x;
    const size_t tid0   = (size_t)blockIdx.x * blockDim.x + threadIdx.x;
    const size_t total  = (size_t)Ntok * DIN;
    for (size_t i = tid0; i < total; i += stride) {
        size_t t = i / DIN;
        int    k = (int)(i - t * DIN);
        float  v = (k < D0) ? X[t * D0 + k] : C[t * DC0 + (k - D0)];
        Xb[i] = (bf16)v;
    }
    for (size_t i = tid0; i < (size_t)n1; i += stride) W1b[i] = (bf16)W1[i];
    for (size_t i = tid0; i < (size_t)n2; i += stride) W2b[i] = (bf16)W2[i];
}

// ---------------------------------------------------------------------------
// Kernel 1: fused GEMM1 + exact GELU + GEMM2 + softmax + top-2.
// One block = 16 tokens, 256 threads = 8 wave32s, each wave owns 384 hidden ch
// in 6 groups of 64 (4 WMMA accumulators share each A-fragment load).
// ---------------------------------------------------------------------------
__global__ __launch_bounds__(256)
void router_fused_kernel(const bf16* __restrict__ Xb,   // [N,1536] bf16
                         const bf16* __restrict__ W1b,  // [3072,1536] bf16
                         const bf16* __restrict__ W2b,  // [64,3072] bf16
                         float* __restrict__ out, int Ntok) {
    __shared__ bf16  sX[TPB * DIN];        // 48 KB: bf16 token tile (TDM target)
    __shared__ bf16  sStage[8][TPB * 32];  // 8 KB : per-wave GELU restage (D->A relayout)
    __shared__ float sL[TPB * NEXP];       // 4 KB : logits / probs
    __shared__ float sTopV[TPB][2];
    __shared__ int   sTopI[TPB][2];

    const int tid = threadIdx.x;
    const int t0  = blockIdx.x * TPB;

    // ---- Stage 1: stage the (pre-concatenated, bf16) token tile into LDS.
    // The 16-token tile is a single contiguous 48 KB span of Xb -> 1-D TDM tile.
#ifdef HAVE_TDM
    if (tid < 32) {   // wave 0 issues one Tensor Data Mover descriptor
        unsigned long long gaddr =
            (unsigned long long)(size_t)(Xb + (size_t)t0 * DIN);
        u32x4 g0;
        g0.x = 1u;                                   // count=1, user descriptor
        g0.y = lds_byte_offset(sX);                  // LDS destination (bytes)
        g0.z = (unsigned int)(gaddr & 0xFFFFFFFFu);  // global_addr[31:0]
        g0.w = (unsigned int)((gaddr >> 32) & 0x01FFFFFFu) | 0x80000000u; // [56:32] | type=2
        const unsigned int tdim0 = (unsigned int)Ntok * DIN;  // 1-D tensor length
        i32x8 g1;
        g1[0] = 0x00010000;                          // data_size = 2 bytes
        g1[1] = (int)((tdim0 & 0xFFFFu) << 16);      // tensor_dim0[15:0]
        g1[2] = (int)(((tdim0 >> 16) & 0xFFFFu) | (1u << 16)); // dim0[31:16] | tensor_dim1=1
        g1[3] = (int)((TPB * DIN) << 16);            // tile_dim0 = 24576
        g1[4] = 0;                                   // tile_dim1 = tile_dim2 = 0
        g1[5] = (int)tdim0;                          // tensor_dim0_stride lo32
        g1[6] = 0;
        g1[7] = 0;
        i32x4 gz4 = {0, 0, 0, 0};
        i32x8 gz8 = {0, 0, 0, 0, 0, 0, 0, 0};
        __builtin_amdgcn_tensor_load_to_lds(g0, g1, gz4, gz4, gz8, 0);
        __builtin_amdgcn_s_wait_tensorcnt(0);
    }
#else
    {
        const v8bf* src = (const v8bf*)(Xb + (size_t)t0 * DIN);
        v8bf*       dst = (v8bf*)sX;
        for (int idx = tid; idx < TPB * DIN / 8; idx += 256) dst[idx] = src[idx];
    }
#endif
    for (int idx = tid; idx < TPB * NEXP; idx += 256) sL[idx] = 0.0f;
    __syncthreads();

    const int wave  = tid >> 5;
    const int lane  = tid & 31;
    const int col   = lane & 15;   // M (A) / N (B,D) lane index
    const int khalf = lane >> 4;   // which K half this lane holds

    v8f zero = {};
    v8f acc2[4] = {zero, zero, zero, zero};   // partial logits, 4 expert tiles x 16

    bf16* stage = &sStage[wave][0];
    const bf16* xrow = &sX[col * DIN + khalf * 8];

    for (int g = 0; g < 6; ++g) {
        const int cbase = wave * 384 + g * 64;   // hidden-channel group base (64 ch)

        __builtin_prefetch(W2b + (size_t)col * HDIM + cbase, 0, 1);

        // ---- GEMM1: 4 subtiles x K=1536; one A-fragment feeds 4 WMMAs.
        v8f acc[4] = {zero, zero, zero, zero};
        const bf16* w1base = W1b + (size_t)(cbase + col) * DIN + khalf * 8;
        #pragma unroll 2
        for (int kt = 0; kt < DIN; kt += 32) {
            V16U a;
            a.h[0] = *(const v8bf*)(xrow + kt);       // K = kt + khalf*8 + 0..7
            a.h[1] = *(const v8bf*)(xrow + kt + 16);  // K = kt + 16 + khalf*8 + 0..7
            #pragma unroll
            for (int st = 0; st < 4; ++st) {
                const bf16* w1row = w1base + (size_t)st * 16 * DIN + kt;
                V16U b;
                b.h[0] = *(const v8bf*)(w1row);       // K = kt + khalf*8 + 0..7
                b.h[1] = *(const v8bf*)(w1row + 16);  // K = kt + 16 + khalf*8 + 0..7
                acc[st] = __builtin_amdgcn_wmma_f32_16x16x32_bf16(
                            false, a.v, false, b.v, (short)0, acc[st], false, false);
            }
        }

        // ---- Two GELU-restage + GEMM2 passes (32 channels each).
        #pragma unroll
        for (int p = 0; p < 2; ++p) {
            #pragma unroll
            for (int q = 0; q < 2; ++q) {
                const int st   = p * 2 + q;
                const int mrow = 8 * khalf;
                #pragma unroll
                for (int r = 0; r < 8; ++r) {
                    float x  = acc[st][r];
                    float gl = 0.5f * x * (1.0f + erff(x * 0.70710678118654752440f));
                    // D layout: token m = r + 8*khalf, channel-in-group = q*16 + col
                    stage[(mrow + r) * 32 + q * 16 + col] = (bf16)gl;
                }
            }
            // Wave-local fence: restage stores must land before A-layout reads.
            asm volatile("s_wait_dscnt 0" ::: "memory");

            V16U a2;
            a2.h[0] = *(const v8bf*)&stage[col * 32 + khalf * 8];
            a2.h[1] = *(const v8bf*)&stage[col * 32 + 16 + khalf * 8];
            const int c2 = cbase + p * 32;
            #pragma unroll
            for (int et = 0; et < 4; ++et) {
                const bf16* w2row = W2b + (size_t)(et * 16 + col) * HDIM + c2 + khalf * 8;
                V16U b2;
                b2.h[0] = *(const v8bf*)(w2row);
                b2.h[1] = *(const v8bf*)(w2row + 16);
                acc2[et] = __builtin_amdgcn_wmma_f32_16x16x32_bf16(
                             false, a2.v, false, b2.v, (short)0, acc2[et], false, false);
            }
            asm volatile("s_wait_dscnt 0" ::: "memory");  // before next restage
        }
    }

    // ---- Cross-wave logit reduction (ds_add_f32 atomics into sL).
    #pragma unroll
    for (int et = 0; et < 4; ++et) {
        #pragma unroll
        for (int r = 0; r < 8; ++r) {
            int m = 8 * khalf + r;
            int e = et * 16 + col;
            atomicAdd(&sL[m * NEXP + e], acc2[et][r]);
        }
    }
    __syncthreads();

    // ---- Softmax (+EPS, clip) and top-2: one thread per token.
    if (tid < TPB) {
        const int m = tid;
        float mx = -3.402823466e38f;
        for (int e = 0; e < NEXP; ++e) mx = fmaxf(mx, sL[m * NEXP + e]);
        float sum = 0.0f;
        for (int e = 0; e < NEXP; ++e) {
            float ex = expf(sL[m * NEXP + e] - mx);
            sL[m * NEXP + e] = ex;
            sum += ex;
        }
        float inv = 1.0f / sum;
        float v1 = -1.0f, v2 = -1.0f;
        int   i1 = 0, i2 = 0;
        for (int e = 0; e < NEXP; ++e) {
            float p = sL[m * NEXP + e] * inv + 1e-9f;
            p = fminf(fmaxf(p, 1e-9f), 1.0f - 1e-9f);
            sL[m * NEXP + e] = p;     // sL now holds clipped probs
            if (p > v1)      { v2 = v1; i2 = i1; v1 = p; i1 = e; }
            else if (p > v2) { v2 = p; i2 = e; }
        }
        sTopV[m][0] = v1; sTopV[m][1] = v2;
        sTopI[m][0] = i1; sTopI[m][1] = i2;
    }
    __syncthreads();

    // ---- Emit all four outputs.
    const size_t N = (size_t)Ntok;
    float* out_mask = out;
    float* out_topi = out + N * NEXP;
    float* out_rp   = out + N * NEXP + N * 2;
    float* out_p    = out + 2 * (N * NEXP) + N * 2;

    for (int idx = tid; idx < TPB * NEXP; idx += 256) {
        int m = idx >> 6, e = idx & 63;
        size_t o = (size_t)(t0 + m) * NEXP + e;
        float p  = sL[idx];
        int   i1 = sTopI[m][0], i2 = sTopI[m][1];
        float v1 = sTopV[m][0], v2 = sTopV[m][1];
        float rs = v1 + v2;
        bool h1 = (e == i1), h2 = (e == i2);
        out_p[o]    = p;
        out_mask[o] = (h1 || h2) ? 1.0f : 0.0f;
        out_rp[o]   = h1 ? (v1 / rs) : (h2 ? (v2 / rs) : 0.0f);
    }
    if (tid < TPB * 2) {
        int m = tid >> 1, k = tid & 1;
        out_topi[(size_t)(t0 + m) * 2 + k] = (float)sTopI[m][k];
    }
}

// ---------------------------------------------------------------------------
extern "C" void kernel_launch(void* const* d_in, const int* in_sizes, int n_in,
                              void* d_out, int out_size, void* d_ws, size_t ws_size,
                              hipStream_t stream) {
    const float* X  = (const float*)d_in[0];  // [B,T,1024]
    const float* C  = (const float*)d_in[1];  // [B,T,512]
    const float* W1 = (const float*)d_in[2];  // [3072,1536]
    const float* W2 = (const float*)d_in[3];  // [64,3072]

    const int n1   = in_sizes[2];             // 3072*1536
    const int n2   = in_sizes[3];             // 64*3072
    const int Ntok = in_sizes[0] / D0;        // B*T = 32768

    bf16* Xb  = (bf16*)d_ws;                  // [N,1536]
    bf16* W1b = Xb + (size_t)Ntok * DIN;      // [3072,1536]
    bf16* W2b = W1b + n1;                     // [64,3072]

    prep_kernel<<<4096, 256, 0, stream>>>(X, C, W1, W2, Xb, W1b, W2b, Ntok, n1, n2);
    router_fused_kernel<<<Ntok / TPB, 256, 0, stream>>>(Xb, W1b, W2b,
                                                        (float*)d_out, Ntok);
}
